// EncoderLSTM_15985868276042
// MI455X (gfx1250) — matrix operand order repaired and checked by reading, not compiled, not run on hardware
//
#include <hip/hip_runtime.h>
#include <hip/hip_bf16.h>
#include <stdint.h>

// Problem constants (match reference)
#define TSTEPS   2048
#define BATCH    64
#define IDIM     256
#define CDIM     256
#define GDIM     (4*CDIM)      // 1024 gate rows (i,f,g,o blocks of 256)
#define KDIM     (IDIM+CDIM)   // 512 fused K (x ; h)
#define NWG      16            // persistent workgroups, each owns 16 cells
#define NTHREADS 256           // 8 wave32 per WG

typedef __attribute__((ext_vector_type(16))) __bf16 v16bf;
typedef __attribute__((ext_vector_type(8)))  float  v8f;

union BF16x16 { v16bf v; uint4 q[2]; };

// ---------------------------------------------------------------------------
// Prep: pack combined weights [G][K] to bf16, fuse biases, broadcast h0,
// reset grid barrier state (re-done every launch: ws is poisoned once and
// never restored between graph replays).
// ---------------------------------------------------------------------------
__global__ void lstm_prep_wb(const float* __restrict__ W_ih,
                             const float* __restrict__ W_hh,
                             const float* __restrict__ b_ih,
                             const float* __restrict__ b_hh,
                             const float* __restrict__ h0,
                             __hip_bfloat16* __restrict__ Wc,
                             float* __restrict__ biasc,
                             __hip_bfloat16* __restrict__ hbuf,
                             unsigned* __restrict__ bar) {
  int idx = blockIdx.x * blockDim.x + threadIdx.x;
  if (idx < GDIM * KDIM) {
    int r = idx / KDIM, k = idx % KDIM;
    float w = (k < IDIM) ? W_ih[r * IDIM + k] : W_hh[r * CDIM + (k - IDIM)];
    Wc[idx] = __float2bfloat16(w);
  }
  if (idx < GDIM) biasc[idx] = b_ih[idx] + b_hh[idx];
  if (idx < BATCH * CDIM) hbuf[idx] = __float2bfloat16(h0[idx % CDIM]); // buf 0
  if (idx < 2) bar[idx] = 0u;
}

// Prep: embs fp32 -> bf16 (~200 MB traffic @ 23.3 TB/s ~ 9 us)
__global__ void lstm_prep_x(const float* __restrict__ embs,
                            __hip_bfloat16* __restrict__ xb, long n) {
  long i = (long)blockIdx.x * blockDim.x + threadIdx.x;
  long stride = (long)gridDim.x * blockDim.x;
  for (; i < n; i += stride) xb[i] = __float2bfloat16(embs[i]);
}

// ---------------------------------------------------------------------------
// Persistent recurrent kernel.
// WG wg owns cells [wg*16, wg*16+16). Wave w: gate = w%4 (i,f,g,o block),
// n-tiles (batch) = {(w/4)*2, (w/4)*2+1}. A = weights 16x32 bf16 fragments,
// resident in 128 VGPRs across all 2048 steps.
//
// Split-barrier pipeline: only the h-half (kt 8..15) of the step GEMM is on
// the inter-WG serial cycle; the x-half (kt 0..7) for step t+1 is computed
// between barrier-arrive and barrier-wait, hidden in the sync shadow.
// ---------------------------------------------------------------------------
__launch_bounds__(NTHREADS)
__global__ void lstm_persistent(const __hip_bfloat16* __restrict__ xb,   // [T][B][I] bf16
                                const __hip_bfloat16* __restrict__ Wc,   // [G][K] bf16
                                const float* __restrict__ biasc,         // [G]
                                const float* __restrict__ c0,            // [C]
                                __hip_bfloat16* __restrict__ hbuf,       // [2][B][C] bf16
                                float* __restrict__ out,                 // [T][B][C]
                                unsigned* __restrict__ bar) {
  __shared__ float gates_lds[4 * 16 * BATCH]; // [gate][cell][batch] 16 KB
  __shared__ float c_lds[16 * BATCH];         // [cell][batch] 4 KB

  unsigned* cnt = bar;
  unsigned* gen = bar + 1;
  const unsigned nb = (unsigned)gridDim.x;

  const int wg   = blockIdx.x;
  const int tid  = threadIdx.x;
  const int wave = tid >> 5;
  const int lane = tid & 31;
  const int gate = wave & 3;
  const int nt0  = (wave >> 2) * 2;
  const int mrow = lane & 15;             // M for A-frag / N (batch col) for B-frag
  const int hi   = (lane >= 16) ? 1 : 0;  // lane-half selects K sub-ranges

  // c state init: reference broadcasts c0[C] over batch
  for (int i = tid; i < 16 * BATCH; i += NTHREADS)
    c_lds[i] = c0[wg * 16 + (i >> 6)];
  __syncthreads();

  // Resident A fragments: 16 k-tiles x (16 bf16 per lane) = 128 VGPRs.
  // 16-bit A 16x32 layout (contiguous, matches B): lanes 0-15 hold K 0..15;
  // lanes 16-31 hold K 16..31; M = lane%16.
  BF16x16 Areg[16];
  {
    const uint4* wrow =
        (const uint4*)(Wc + (size_t)(gate * 256 + wg * 16 + mrow) * KDIM);
#pragma unroll
    for (int kt = 0; kt < 16; ++kt) {
      Areg[kt].q[0] = wrow[kt * 4 + hi * 2];       // K = kt*32 + hi*16 .. +7
      Areg[kt].q[1] = wrow[kt * 4 + hi * 2 + 1];   // K = kt*32 + hi*16 + 8 .. +15
    }
  }

  // Fused bias (b_ih+b_hh); C/D layout: VGPR j -> row j + hi*8, col lane%16.
  float biasv[8];
#pragma unroll
  for (int j = 0; j < 8; ++j)
    biasv[j] = biasc[gate * 256 + wg * 16 + j + hi * 8];

  const int b0 = nt0 * 16 + mrow;   // batch column, n-tile 0
  const int b1 = b0 + 16;           // batch column, n-tile 1

  // x-half of the step GEMM (kt 0..7): depends only on precomputed x.
  // 16-bit B 32x16 order used here (contiguous, matches A): lanes 0-15 hold K kt*32+0..15 of column
  // lane%16; lanes 16-31 hold K kt*32+16..31 -> one 32B contiguous run.
  auto xpart = [&](const __hip_bfloat16* xt, v8f& a0, v8f& a1) {
#pragma unroll
    for (int j = 0; j < 8; ++j) { a0[j] = biasv[j]; a1[j] = biasv[j]; }
#pragma unroll
    for (int kt = 0; kt < 8; ++kt) {
      int ko = kt * 32 + hi * 16;
      const uint4* p0 = (const uint4*)(xt + (size_t)b0 * IDIM + ko);
      const uint4* p1 = (const uint4*)(xt + (size_t)b1 * IDIM + ko);
      BF16x16 Bf0, Bf1;
      Bf0.q[0] = p0[0]; Bf0.q[1] = p0[1];
      Bf1.q[0] = p1[0]; Bf1.q[1] = p1[1];
      a0 = __builtin_amdgcn_wmma_f32_16x16x32_bf16(
          false, Areg[kt].v, false, Bf0.v, (short)0, a0, false, false);
      a1 = __builtin_amdgcn_wmma_f32_16x16x32_bf16(
          false, Areg[kt].v, false, Bf1.v, (short)0, a1, false, false);
    }
  };

  v8f acc0, acc1;
  xpart(xb, acc0, acc1);            // prologue: x contribution of step 0

  for (int t = 0; t < TSTEPS; ++t) {
    const __hip_bfloat16* hc = hbuf + (size_t)(t & 1) * BATCH * CDIM;
    __hip_bfloat16* hn = hbuf + (size_t)((t & 1) ^ 1) * BATCH * CDIM;

    // --- h-half (critical path): kt 8..15 over the fresh hidden state ---
#pragma unroll
    for (int kt = 0; kt < 8; ++kt) {
      int ko = kt * 32 + hi * 16;   // offset within h's 256 columns
      const uint4* p0 = (const uint4*)(hc + (size_t)b0 * CDIM + ko);
      const uint4* p1 = (const uint4*)(hc + (size_t)b1 * CDIM + ko);
      BF16x16 Bf0, Bf1;
      Bf0.q[0] = p0[0]; Bf0.q[1] = p0[1];
      Bf1.q[0] = p1[0]; Bf1.q[1] = p1[1];
      acc0 = __builtin_amdgcn_wmma_f32_16x16x32_bf16(
          false, Areg[8 + kt].v, false, Bf0.v, (short)0, acc0, false, false);
      acc1 = __builtin_amdgcn_wmma_f32_16x16x32_bf16(
          false, Areg[8 + kt].v, false, Bf1.v, (short)0, acc1, false, false);
    }

    // Scatter gate tiles to LDS: VGPR j of C/D holds (row j+hi*8, col lane%16).
#pragma unroll
    for (int j = 0; j < 8; ++j) {
      int cell = j + hi * 8;
      gates_lds[gate * (16 * BATCH) + cell * BATCH + b0] = acc0[j];
      gates_lds[gate * (16 * BATCH) + cell * BATCH + b1] = acc1[j];
    }
    __syncthreads();

    // Elementwise cell update: 16 cells x 64 batch = 1024 items / 256 threads
#pragma unroll
    for (int i2 = 0; i2 < 4; ++i2) {
      int idx  = tid + i2 * NTHREADS;  // cell*64 + b
      int cell = idx >> 6;
      int b    = idx & 63;
      float ig = gates_lds[0 * 1024 + idx];
      float fg = gates_lds[1 * 1024 + idx];
      float gg = gates_lds[2 * 1024 + idx];
      float og = gates_lds[3 * 1024 + idx];
      float is = 1.f / (1.f + __expf(-ig));
      float fs = 1.f / (1.f + __expf(-fg));
      float gt = tanhf(gg);
      float os = 1.f / (1.f + __expf(-og));
      float cn = fs * c_lds[idx] + is * gt;
      float hv = os * tanhf(cn);
      c_lds[idx] = cn;
      int cg = wg * 16 + cell;
      out[(size_t)t * (BATCH * CDIM) + (size_t)b * CDIM + cg] = hv;
      hn[b * CDIM + cg] = __float2bfloat16(hv);
    }

    // --- split grid barrier: arrive, overlap x-half of t+1, then wait ---
    __threadfence();                 // publish h/out to device scope
    __syncthreads();                 // all threads' stores issued
    unsigned g = 0;
    if (tid == 0) {
      g = __atomic_load_n(gen, __ATOMIC_RELAXED);
      if (atomicAdd(cnt, 1u) == nb - 1u) {   // last WG: reset + release
        __atomic_store_n(cnt, 0u, __ATOMIC_RELAXED);
        __threadfence();
        __atomic_store_n(gen, g + 1u, __ATOMIC_RELEASE);
      }
    }

    if (t + 1 < TSTEPS) {            // hidden in the barrier-wait shadow
      xpart(xb + (size_t)(t + 1) * BATCH * IDIM, acc0, acc1);
      if (t + 2 < TSTEPS)            // warm L2/WGP$ for the step after
        __builtin_prefetch(xb + (size_t)(t + 2) * BATCH * IDIM + tid * 64, 0, 1);
    }

    if (tid == 0) {
      while (__atomic_load_n(gen, __ATOMIC_ACQUIRE) == g)
        __builtin_amdgcn_s_sleep(1);
      __threadfence();               // acquire other WGs' h writes
    }
    __syncthreads();
  }
}

// ---------------------------------------------------------------------------
// Workspace layout:
//   [0, 1MB)        Wc   : combined weights bf16 [1024][512]
//   [1MB, +4KB)     bias : b_ih + b_hh fp32 [1024]
//   [1MB+4KB,+64KB) hbuf : bf16 [2][64][256] double-buffered h
//   [1MB+68KB,+8B)  bar  : grid barrier {count, gen}
//   [2MB, +64MB)    xb   : bf16 embs [2048][64][256]
// ---------------------------------------------------------------------------
extern "C" void kernel_launch(void* const* d_in, const int* in_sizes, int n_in,
                              void* d_out, int out_size, void* d_ws, size_t ws_size,
                              hipStream_t stream) {
  (void)in_sizes; (void)n_in; (void)out_size; (void)ws_size;
  const float* embs = (const float*)d_in[0];
  const float* W_ih = (const float*)d_in[1];
  const float* W_hh = (const float*)d_in[2];
  const float* b_ih = (const float*)d_in[3];
  const float* b_hh = (const float*)d_in[4];
  const float* h0   = (const float*)d_in[5];
  const float* c0   = (const float*)d_in[6];
  float* out = (float*)d_out;

  char* ws = (char*)d_ws;
  __hip_bfloat16* Wc   = (__hip_bfloat16*)(ws);
  float*          bias = (float*)(ws + (1u << 20));
  __hip_bfloat16* hbuf = (__hip_bfloat16*)(ws + (1u << 20) + 4096);
  unsigned*       bar  = (unsigned*)(ws + (1u << 20) + 4096 + 65536);
  __hip_bfloat16* xb   = (__hip_bfloat16*)(ws + (2u << 20));

  lstm_prep_wb<<<(GDIM * KDIM + 255) / 256, 256, 0, stream>>>(
      W_ih, W_hh, b_ih, b_hh, h0, Wc, bias, hbuf, bar);
  lstm_prep_x<<<2048, 256, 0, stream>>>(
      embs, xb, (long)TSTEPS * BATCH * IDIM);
  lstm_persistent<<<NWG, NTHREADS, 0, stream>>>(
      xb, Wc, bias, c0, hbuf, out, bar);
}
